// Model_47691316855137
// MI455X (gfx1250) — matrix operand order repaired and checked by reading, not compiled, not run on hardware
//
#include <hip/hip_runtime.h>
#include <hip/hip_bf16.h>

typedef __attribute__((ext_vector_type(16))) _Float16 v16h;
typedef __attribute__((ext_vector_type(8)))  _Float16 v8h;
typedef __attribute__((ext_vector_type(4)))  _Float16 v4h;
typedef __attribute__((ext_vector_type(8)))  float    v8f;

#define SNN_BETA 0.95f

__device__ __forceinline__ v8f wmma_f16(v16h a, v16h b, v8f c) {
    // D = A(16x32 f16) * B(32x16 f16) + C(16x16 f32)
    return __builtin_amdgcn_wmma_f32_16x16x32_f16(false, a, false, b, (short)0, c,
                                                  false, false);
}

// ---------------------------------------------------------------------------
// Fully fused persistent SNN: one block per image, 100 timesteps in-kernel.
// The recurrence (mem1/mem2/mem3) is elementwise per image -> no cross-block
// dependency at all. Weights staged to LDS once; membrane state in registers;
// spikes never leave LDS. Global traffic = streaming x + tiny output spikes.
//
// conv1: implicit GEMM M=784(49 tiles) K=50->64 N=12->16   (v_wmma f16)
// conv2: implicit GEMM M=100(7 tiles)  K=300->320 N=32     (v_wmma f16)
// fc:    800x10 per image, VALU dot + LDS reduce (negligible FLOPs)
//
// A-frag element i <-> K = kc*32 + 16*half + i                 (gather)
// B-frag element i <-> K = kc*32 + 16*half + i                 (contiguous b128)
// C/D: lane = N (lane&15), VGPR r = row 8*(lane>>4) + r
// ---------------------------------------------------------------------------
__global__ __launch_bounds__(256)
void snn_forward(const float* __restrict__ x, const float* __restrict__ W1,
                 const float* __restrict__ b1, const float* __restrict__ W2,
                 const float* __restrict__ b2, const float* __restrict__ W3,
                 const float* __restrict__ b3, float* __restrict__ out)
{
    __shared__ alignas(32) _Float16 sWt1[16 * 64];    // conv1 weight panel [N][Kpad]
    __shared__ alignas(32) _Float16 sWt2[32 * 320];   // conv2 weight panel [N][Kpad]
    __shared__ alignas(32) _Float16 sW3[10 * 800];    // fc weights f16
    __shared__ alignas(32) _Float16 sX[3072];         // image [2][32][32] + zero pad
    __shared__ alignas(32) _Float16 sS1[2608];        // spk1 [12][14][14] + zero pad
    __shared__ alignas(32) _Float16 sS2[800];         // spk2 [32][5][5]
    __shared__ int   sOff1[64];                       // conv1 k -> input offset
    __shared__ int   sOff2[320];                      // conv2 k -> input offset
    __shared__ float sConv[12 * 784];  // time-multiplexed: conv1 [12][784] / conv2 [32][100]
    __shared__ float sFc[80];
    __shared__ float sB1[12], sB2[32], sB3[10];

    const int tid = threadIdx.x;
    const int b   = blockIdx.x;

    // ---------------- one-time staging (amortized over 100 steps) ----------------
    for (int e = tid; e < 16 * 64; e += 256) {
        int n = e >> 6, k = e & 63;
        sWt1[e] = (_Float16)((n < 12 && k < 50) ? W1[n * 50 + k] : 0.f);
    }
    for (int e = tid; e < 32 * 320; e += 256) {
        int n = e / 320, k = e % 320;
        sWt2[e] = (_Float16)((k < 300) ? W2[n * 300 + k] : 0.f);
    }
    for (int e = tid; e < 10 * 800; e += 256) sW3[e] = (_Float16)W3[e];
    if (tid < 64) {
        int k = tid;
        if (k < 50) {
            int c = k / 25, rem = k % 25;
            sOff1[k] = c * 1024 + (rem / 5) * 32 + (rem % 5);
        } else {
            sOff1[k] = 2048;          // zero zone: base <= 891 -> idx in [2048,3071]
        }
    }
    for (int k = tid; k < 320; k += 256) {
        if (k < 300) {
            int c = k / 25, rem = k % 25;
            sOff2[k] = c * 196 + (rem / 5) * 14 + (rem % 5);
        } else {
            sOff2[k] = 2352;          // zero zone: base <= 135 -> idx in [2352,2607]
        }
    }
    if (tid < 12) sB1[tid] = b1[tid];
    if (tid < 32) sB2[tid] = b2[tid];
    if (tid < 10) sB3[tid] = b3[tid];
    if (tid < 128) { v8h z = {}; *(v8h*)&sX[2048 + tid * 8] = z; }   // constant pads
    if (tid < 32)  { v8h z = {}; *(v8h*)&sS1[2352 + tid * 8] = z; }
    __syncthreads();

    const int lane = tid & 31;
    const int wave = tid >> 5;
    const int half = lane >> 4;
    const int mrow = lane & 15;
    const int ncol = lane & 15;

    // conv1 B fragments + gather offsets: invariant over time, keep in registers.
    const v16h bf0 = *(const v16h*)&sWt1[ncol * 64 +      half * 16];
    const v16h bf1 = *(const v16h*)&sWt1[ncol * 64 + 32 + half * 16];
    int offA0[16], offA1[16];
    #pragma unroll
    for (int i = 0; i < 16; ++i) {
        int kb = half * 16 + i;
        offA0[i] = sOff1[kb];
        offA1[i] = sOff1[32 + kb];
    }

    // Membrane state in registers (fixed thread<->element mapping per timestep).
    float m1[10], m2[4], m3 = 0.f;
    #pragma unroll
    for (int j = 0; j < 10; ++j) m1[j] = 0.f;
    #pragma unroll
    for (int j = 0; j < 4; ++j) m2[j] = 0.f;

    const size_t xBlock = (size_t)b * 2048;

    for (int t = 0; t < 100; ++t) {
        // ---- stage image t (f32->f16) + prefetch image t+1 into L2 ----
        const float4* xi4 = (const float4*)(x + (size_t)t * 524288 + xBlock);
        for (int q = tid; q < 512; q += 256) {
            float4 f = xi4[q];
            v4h h;
            h[0] = (_Float16)f.x; h[1] = (_Float16)f.y;
            h[2] = (_Float16)f.z; h[3] = (_Float16)f.w;
            *(v4h*)&sX[q * 4] = h;
            if ((q & 7) == 0)
                __builtin_prefetch(xi4 + 131072 + q, 0, 1);  // speculative, 128B lines
        }
        __syncthreads();

        // ---- conv1 implicit GEMM (49 M-tiles over 8 waves) ----
        for (int mt = wave; mt < 49; mt += 8) {
            int p    = mt * 16 + mrow;
            int oy   = p / 28, ox = p % 28;
            int base = oy * 32 + ox;
            v8f acc = {};
            v16h a0, a1;
            #pragma unroll
            for (int i = 0; i < 16; ++i) a0[i] = sX[base + offA0[i]];
            acc = wmma_f16(a0, bf0, acc);
            #pragma unroll
            for (int i = 0; i < 16; ++i) a1[i] = sX[base + offA1[i]];
            acc = wmma_f16(a1, bf1, acc);
            if (ncol < 12) {
                #pragma unroll
                for (int r = 0; r < 8; ++r)
                    sConv[ncol * 784 + mt * 16 + half * 8 + r] = acc[r];
            }
        }
        __syncthreads();

        // ---- maxpool 2x2 + bias + LIF1 -> spikes in LDS ----
        #pragma unroll
        for (int j = 0; j < 10; ++j) {
            int i = tid + 256 * j;
            if (i < 2352) {
                int c = i / 196, rem = i % 196;
                int py = rem / 14, px = rem % 14;
                const float* r0 = &sConv[c * 784 + (2 * py) * 28 + 2 * px];
                float v   = fmaxf(fmaxf(r0[0], r0[1]), fmaxf(r0[28], r0[29]));
                float cur = v + sB1[c];
                float mm  = m1[j];
                float rst = (mm > 1.f) ? 1.f : 0.f;
                mm = SNN_BETA * mm + cur - rst;
                sS1[i] = (_Float16)((mm > 1.f) ? 1.f : 0.f);
                m1[j] = mm;
            }
        }
        __syncthreads();

        // ---- conv2 implicit GEMM (7 M-tiles, one per wave; A reused for 2 N-tiles) ----
        if (wave < 7) {
            const int mt = wave;
            int p = mt * 16 + mrow;
            if (p > 99) p = 99;                       // clamp pad rows (writes guarded)
            int oy = p / 10, ox = p % 10;
            int base = oy * 14 + ox;
            v8f acc0 = {}, acc1 = {};
            for (int kc = 0; kc < 10; ++kc) {
                v16h a;
                #pragma unroll
                for (int i = 0; i < 16; ++i) {
                    int k = kc * 32 + half * 16 + i;
                    a[i] = sS1[base + sOff2[k]];
                }
                const v16h c0 = *(const v16h*)&sWt2[ ncol       * 320 + kc * 32 + half * 16];
                const v16h c1 = *(const v16h*)&sWt2[(16 + ncol) * 320 + kc * 32 + half * 16];
                acc0 = wmma_f16(a, c0, acc0);
                acc1 = wmma_f16(a, c1, acc1);
            }
            #pragma unroll
            for (int r = 0; r < 8; ++r) {
                int pr = mt * 16 + half * 8 + r;
                if (pr < 100) {
                    sConv[ ncol       * 100 + pr] = acc0[r];   // aliased as [32][100]
                    sConv[(16 + ncol) * 100 + pr] = acc1[r];
                }
            }
        }
        __syncthreads();

        // ---- maxpool 2x2 + bias + LIF2 -> spikes in LDS ----
        #pragma unroll
        for (int j = 0; j < 4; ++j) {
            int i = tid + 256 * j;
            if (i < 800) {
                int c = i / 25, rem = i % 25;
                int py = rem / 5, px = rem % 5;
                const float* r0 = &sConv[c * 100 + (2 * py) * 10 + 2 * px];
                float v   = fmaxf(fmaxf(r0[0], r0[1]), fmaxf(r0[10], r0[11]));
                float cur = v + sB2[c];
                float mm  = m2[j];
                float rst = (mm > 1.f) ? 1.f : 0.f;
                mm = SNN_BETA * mm + cur - rst;
                sS2[i] = (_Float16)((mm > 1.f) ? 1.f : 0.f);
                m2[j] = mm;
            }
        }
        __syncthreads();

        // ---- fc partial dot products: 10 classes x 8 partials of 100 ----
        if (tid < 80) {
            int n = tid >> 3, part = tid & 7;
            const _Float16* wr = &sW3[n * 800 + part * 100];
            const _Float16* sr = &sS2[part * 100];
            float s = 0.f;
            #pragma unroll 4
            for (int k = 0; k < 100; ++k) s += (float)sr[k] * (float)wr[k];
            sFc[tid] = s;
        }
        __syncthreads();

        // ---- reduce + LIF3 -> output spikes ----
        if (tid < 10) {
            float cur = sB3[tid];
            #pragma unroll
            for (int p = 0; p < 8; ++p) cur += sFc[tid * 8 + p];
            float mm  = m3;
            float rst = (mm > 1.f) ? 1.f : 0.f;
            mm = SNN_BETA * mm + cur - rst;
            m3 = mm;
            out[(size_t)t * 2560 + b * 10 + tid] = (mm > 1.f) ? 1.f : 0.f;
        }
        __syncthreads();
    }
}

// ---------------------------------------------------------------------------
// Host driver: a single persistent kernel launch. No workspace needed.
// ---------------------------------------------------------------------------
extern "C" void kernel_launch(void* const* d_in, const int* in_sizes, int n_in,
                              void* d_out, int out_size, void* d_ws, size_t ws_size,
                              hipStream_t stream) {
    (void)in_sizes; (void)n_in; (void)out_size; (void)d_ws; (void)ws_size;

    const float* x  = (const float*)d_in[0];   // [100,256,2,32,32]
    const float* W1 = (const float*)d_in[1];   // [12,2,5,5]
    const float* b1 = (const float*)d_in[2];   // [12]
    const float* W2 = (const float*)d_in[3];   // [32,12,5,5]
    const float* b2 = (const float*)d_in[4];   // [32]
    const float* W3 = (const float*)d_in[5];   // [10,800]
    const float* b3 = (const float*)d_in[6];   // [10]
    float*      out = (float*)d_out;           // [100,256,10]

    snn_forward<<<256, 256, 0, stream>>>(x, W1, b1, W2, b2, W3, b3, out);
}
